// Informer5_5_575525618094
// MI455X (gfx1250) — compile-verified, matrix operand order checked by reading, not run on hardware
//
#include <hip/hip_runtime.h>
#include <hip/hip_bf16.h>

// ---------------------------------------------------------------------------
// Informer pipeline for MI455X (gfx1250, wave32).
// Memory-bound on FFN weights (~1.36 GB fp32/call @ 23.3 TB/s ~ 58us floor).
// GEMMs: v_wmma_f32_16x16x32_bf16 (fp32 -> bf16 in-register).
// LDS staging: global_load_async_to_lds_b128 + s_wait_asynccnt (CDNA5 async).
// ---------------------------------------------------------------------------

#define SEQL   128
#define BATCH  32
#define DMODEL 32
#define NHEADS 8
#define HEADD  4
#define UTOP   25

typedef __attribute__((ext_vector_type(16))) __bf16 v16bf;
typedef __attribute__((ext_vector_type(8)))  float  v8f;

// Flat LDS pointer -> wave-relative LDS byte offset (low 32 bits of the pointer).
static __device__ __forceinline__ unsigned lds_off(const void* p) {
  return (unsigned)(size_t)p;
}

// Async copy 16 bytes global -> LDS, bypassing VGPRs (tracked by ASYNCcnt).
static __device__ __forceinline__ void async_g2l_b128(unsigned loff, const void* g) {
#if defined(__HIP_DEVICE_COMPILE__)
  asm volatile("global_load_async_to_lds_b128 %0, %1, off"
               :: "v"(loff), "v"(g) : "memory");
#endif
}
static __device__ __forceinline__ void wait_async0() {
#if defined(__HIP_DEVICE_COMPILE__)
  asm volatile("s_wait_asynccnt 0x0" ::: "memory");
#endif
}

// Per-lane fragment loader: 8 contiguous fp32 at p0 -> elements 0..7,
// 8 contiguous fp32 at p1 -> elements 8..15 (K = g*8+e and 16+g*8+e layout).
static __device__ __forceinline__ v16bf load_frag_contig(const float* p0, const float* p1) {
  const float4* a = (const float4*)p0;
  const float4* b = (const float4*)p1;
  float4 x0 = a[0], x1 = a[1];
  float4 y0 = b[0], y1 = b[1];
  v16bf f;
  f[0]=(__bf16)x0.x; f[1]=(__bf16)x0.y; f[2]=(__bf16)x0.z; f[3]=(__bf16)x0.w;
  f[4]=(__bf16)x1.x; f[5]=(__bf16)x1.y; f[6]=(__bf16)x1.z; f[7]=(__bf16)x1.w;
  f[8]=(__bf16)y0.x; f[9]=(__bf16)y0.y; f[10]=(__bf16)y0.z; f[11]=(__bf16)y0.w;
  f[12]=(__bf16)y1.x; f[13]=(__bf16)y1.y; f[14]=(__bf16)y1.z; f[15]=(__bf16)y1.w;
  return f;
}

static __device__ __forceinline__ v8f wmma_bf16(v16bf a, v16bf b, v8f c) {
  return __builtin_amdgcn_wmma_f32_16x16x32_bf16(false, a, false, b, (short)0, c, false, false);
}

// ---------------------------------------------------------------------------
// Conv tower: 8 branches, 3 conv1d layers each, summed.  One WG per batch.
// Input row staged into LDS with async-to-LDS B128 (one per thread).
// ---------------------------------------------------------------------------
struct ConvP  { const float *w1,*b1,*w2,*b2,*w3,*b3; int k; };
struct ConvAll { ConvP p[8]; };

__global__ __launch_bounds__(256) void conv_kernel(const float* __restrict__ x,
                                                   ConvAll all,
                                                   float* __restrict__ s) {
  int b = blockIdx.x, tid = threadIdx.x;
  __shared__ __align__(16) float xs[1024];
  __shared__ float y1[4*512];
  __shared__ float y2[16*256];
  __shared__ float acc[32*128];
  // 256 threads x 16B = 4KB row, memory -> LDS without touching VGPRs
  async_g2l_b128(lds_off(&xs[tid*4]), x + (size_t)b*1024 + tid*4);
  for (int i=tid;i<4096;i+=256) acc[i] = 0.f;
  wait_async0();
  __syncthreads();
  for (int blk=0; blk<8; blk++){
    ConvP P = all.p[blk];
    int kk = P.k, pad = (kk-2)>>1;
    // conv1: 1->4 ch, stride 2, L 1024->512, relu
    for (int i=tid;i<4*512;i+=256){
      int oc=i>>9, op=i&511;
      float a = P.b1[oc];
      int st = op*2 - pad;
      const float* w = P.w1 + oc*kk;
      for (int t=0;t<kk;t++){ int ip=st+t; if ((unsigned)ip < 1024u) a += w[t]*xs[ip]; }
      y1[i] = fmaxf(a, 0.f);
    }
    __syncthreads();
    // conv2: 4->16 ch, k=24 pad=11 stride 2, L 512->256, relu
    for (int i=tid;i<16*256;i+=256){
      int oc=i>>8, op=i&255;
      float a = P.b2[oc]; int st=op*2-11;
      for (int ic=0;ic<4;ic++){
        const float* w  = P.w2 + (oc*4+ic)*24;
        const float* yy = y1 + ic*512;
        #pragma unroll
        for (int t=0;t<24;t++){ int ip=st+t; if((unsigned)ip<512u) a += w[t]*yy[ip]; }
      }
      y2[i]=fmaxf(a,0.f);
    }
    __syncthreads();
    // conv3: 16->32 ch, k=24 pad=11 stride 2, L 256->128, relu, accumulate
    for (int i=tid;i<32*128;i+=256){
      int oc=i>>7, op=i&127;
      float a = P.b3[oc]; int st=op*2-11;
      for (int ic=0;ic<16;ic++){
        const float* w  = P.w3 + (oc*16+ic)*24;
        const float* yy = y2 + ic*256;
        #pragma unroll
        for (int t=0;t<24;t++){ int ip=st+t; if((unsigned)ip<256u) a += w[t]*yy[ip]; }
      }
      acc[i] += fmaxf(a,0.f);
    }
    __syncthreads();
  }
  for (int i=tid;i<4096;i+=256) s[(size_t)b*4096 + i] = acc[i];
}

// ---------------------------------------------------------------------------
// QKV projection: rows r = l*32+b (M=4096), K=N=32.  One wave per M-tile,
// blockIdx.y selects q/k/v.  A gathered from (B,C,L) layout; W rows contiguous.
// ---------------------------------------------------------------------------
__global__ __launch_bounds__(32) void qkv_kernel(const float* __restrict__ x,
    const float* __restrict__ wq, const float* __restrict__ bq,
    const float* __restrict__ wk, const float* __restrict__ bk,
    const float* __restrict__ wv, const float* __restrict__ bv,
    float* __restrict__ q, float* __restrict__ k, float* __restrict__ v) {
  int lane = threadIdx.x;
  int mtile = blockIdx.x;
  int which = blockIdx.y;
  const float* W   = (which==0)?wq:(which==1)?wk:wv;
  const float* bia = (which==0)?bq:(which==1)?bk:bv;
  float* outp      = (which==0)?q :(which==1)?k :v;
  int ml = lane & 15, g = lane >> 4;
  int r = mtile*16 + ml;
  int l = r >> 5, b = r & 31;
  const float* xb = x + (size_t)b*(DMODEL*SEQL) + l;
  v16bf a;
  #pragma unroll
  for (int e=0;e<8;e++) a[e]   = (__bf16)xb[(g*8+e)*SEQL];
  #pragma unroll
  for (int e=0;e<8;e++) a[8+e] = (__bf16)xb[(16+g*8+e)*SEQL];
  const float* w0 = W + (size_t)ml*DMODEL;
  const float* w1 = W + (size_t)(16+ml)*DMODEL;
  v16bf bf0 = load_frag_contig(w0 + g*8, w0 + 16 + g*8);
  v16bf bf1 = load_frag_contig(w1 + g*8, w1 + 16 + g*8);
  v8f c0 = {}; v8f c1 = {};
  c0 = wmma_bf16(a, bf0, c0);
  c1 = wmma_bf16(a, bf1, c1);
  int nl = lane & 15;
  #pragma unroll
  for (int rr=0; rr<8; rr++){
    int mrow = (lane < 16) ? rr : (8 + rr);
    int rg = mtile*16 + mrow;
    int ll = rg >> 5, bb = rg & 31;
    int n0 = nl, n1 = 16 + nl;
    float v0 = c0[rr] + bia[n0];
    float v1 = c1[rr] + bia[n1];
    outp[(((size_t)bb*NHEADS + (n0>>2))*SEQL + ll)*HEADD + (n0&3)] = v0;
    outp[(((size_t)bb*NHEADS + (n1>>2))*SEQL + ll)*HEADD + (n1&3)] = v1;
  }
}

// ---------------------------------------------------------------------------
// ProbSparse attention: one WG of 128 threads per (b,h).  q/k/v rows staged
// into LDS via async-to-LDS B128 (one 16B row per thread per tensor).
// ---------------------------------------------------------------------------
__global__ __launch_bounds__(128) void prob_attn_kernel(const float* __restrict__ q,
    const float* __restrict__ k, const float* __restrict__ v,
    float* __restrict__ ob, int layer) {
  int bh = blockIdx.x; int b = bh >> 3, h = bh & 7;
  int tid = threadIdx.x;
  __shared__ __align__(16) float qs[SEQL][HEADD];
  __shared__ __align__(16) float ks[SEQL][HEADD];
  __shared__ __align__(16) float vs[SEQL][HEADD];
  __shared__ __align__(16) float ctx[SEQL][HEADD];
  __shared__ float red[SEQL];
  __shared__ int   redi[SEQL];
  __shared__ float Ms[SEQL];
  __shared__ int   topidx[UTOP];
  {
    size_t base = ((size_t)bh*SEQL + tid)*HEADD;
    async_g2l_b128(lds_off(&qs[tid][0]), q + base);
    async_g2l_b128(lds_off(&ks[tid][0]), k + base);
    async_g2l_b128(lds_off(&vs[tid][0]), v + base);
  }
  wait_async0();
  __syncthreads();
  // sparsity measure M[l] = max_s qk - mean over sampled keys
  float mx = -3.0e38f, sm = 0.f;
  for (int s=0;s<UTOP;s++){
    unsigned hsh = 0x9E3779B9u*(unsigned)(layer+1)
                 ^ (unsigned)(tid)*2654435761u
                 ^ ((unsigned)(s)*40503u + 0x7feb352du);
    hsh ^= hsh>>15; hsh *= 0x2c1b3c6du; hsh ^= hsh>>12;
    int j = (int)(hsh & (SEQL-1));
    float d0 = qs[tid][0]*ks[j][0]+qs[tid][1]*ks[j][1]
             + qs[tid][2]*ks[j][2]+qs[tid][3]*ks[j][3];
    mx = fmaxf(mx, d0); sm += d0;
  }
  Ms[tid] = mx - sm*(1.0f/SEQL);
  __syncthreads();
  // top-u selection (argmax tree, tie -> lower index, like top_k)
  for (int t=0;t<UTOP;t++){
    red[tid]=Ms[tid]; redi[tid]=tid; __syncthreads();
    for (int off=SEQL/2; off>0; off>>=1){
      if (tid < off){
        float o = red[tid+off]; int oi = redi[tid+off];
        if (o > red[tid] || (o == red[tid] && oi < redi[tid])) { red[tid]=o; redi[tid]=oi; }
      }
      __syncthreads();
    }
    if (tid==0){ topidx[t]=redi[0]; Ms[redi[0]] = -3.0e38f; }
    __syncthreads();
  }
  // ctx = inclusive cumsum of v
  #pragma unroll
  for (int d=0; d<HEADD; d++){
    float a=0.f;
    for (int j=0;j<=tid;j++) a += vs[j][d];
    ctx[tid][d]=a;
  }
  __syncthreads();
  // attention for the selected query rows (causal, scale = 1/sqrt(4))
  for (int t=0;t<UTOP;t++){
    int row = topidx[t];
    float sc = (tid > row) ? -3.0e38f
             : 0.5f*(qs[row][0]*ks[tid][0]+qs[row][1]*ks[tid][1]
                    +qs[row][2]*ks[tid][2]+qs[row][3]*ks[tid][3]);
    red[tid]=sc; __syncthreads();
    for (int off=SEQL/2; off>0; off>>=1){
      if (tid<off) red[tid]=fmaxf(red[tid],red[tid+off]);
      __syncthreads();
    }
    float smax = red[0]; __syncthreads();
    float e = (tid>row)?0.f:__expf(sc - smax);
    red[tid]=e; __syncthreads();
    for (int off=SEQL/2; off>0; off>>=1){
      if (tid<off) red[tid]+=red[tid+off];
      __syncthreads();
    }
    float denom = red[0]; __syncthreads();
    float attn = e/denom;
    #pragma unroll
    for (int d=0; d<HEADD; d++){
      red[tid]=attn*vs[tid][d]; __syncthreads();
      for (int off=SEQL/2; off>0; off>>=1){
        if (tid<off) red[tid]+=red[tid+off];
        __syncthreads();
      }
      if (tid==0) ctx[row][d]=red[0];
      __syncthreads();
    }
  }
  // write (L,B,DMODEL) rows for the output projection GEMM
  float* op = ob + ((size_t)tid*BATCH + b)*DMODEL + h*HEADD;
  #pragma unroll
  for (int d=0; d<HEADD; d++) op[d]=ctx[tid][d];
}

// ---------------------------------------------------------------------------
// Attention output projection + relu + residual + BN1, writing (B,C,L).
// ---------------------------------------------------------------------------
__global__ __launch_bounds__(32) void attnproj_kernel(const float* __restrict__ ao,
    const float* __restrict__ wo, const float* __restrict__ bo,
    const float* __restrict__ xold, const float* __restrict__ g1,
    const float* __restrict__ b1, float* __restrict__ xnew) {
  int lane = threadIdx.x;
  int mtile = blockIdx.x;
  int ml = lane & 15, g = lane >> 4;
  const float* arow = ao + (size_t)(mtile*16 + ml)*DMODEL;
  v16bf a = load_frag_contig(arow + g*8, arow + 16 + g*8);
  const float* w0 = wo + (size_t)ml*DMODEL;
  const float* w1 = wo + (size_t)(16+ml)*DMODEL;
  v16bf bf0 = load_frag_contig(w0+g*8, w0+16+g*8);
  v16bf bf1 = load_frag_contig(w1+g*8, w1+16+g*8);
  v8f c0={}, c1={};
  c0 = wmma_bf16(a, bf0, c0);
  c1 = wmma_bf16(a, bf1, c1);
  const float inv = rsqrtf(1.0f + 1e-5f);
  int nl = lane & 15;
  #pragma unroll
  for (int rr=0;rr<8;rr++){
    int mrow = (lane<16)?rr:(8+rr);
    int rg = mtile*16 + mrow;
    int ll = rg>>5, bb = rg&31;
    #pragma unroll
    for (int half=0; half<2; half++){
      int n = half*16 + nl;
      float val = (half? c1[rr] : c0[rr]) + bo[n];
      val = fmaxf(val, 0.f);
      size_t oi = (size_t)bb*(DMODEL*SEQL) + (size_t)n*SEQL + ll;
      xnew[oi] = (val + xold[oi]) * (g1[n]*inv) + b1[n];
    }
  }
}

// ---------------------------------------------------------------------------
// Streaming M=32 GEMM: out(32,N) = A(32,K) @ W(N,K)^T + bias, optional relu
// and optional (resid + BN over channel n>>7).  One wave per 16-col tile,
// two accumulators cover all 32 rows; W rows streamed once from HBM with
// prefetch; A stays L2-resident.
// ---------------------------------------------------------------------------
template<bool RELU, bool BN>
__global__ __launch_bounds__(256) void wmma_gemm32(const float* __restrict__ A,
    const float* __restrict__ W, const float* __restrict__ bias,
    const float* __restrict__ resid, const float* __restrict__ bng,
    const float* __restrict__ bnb, float* __restrict__ out, int K, int N) {
  int lane = threadIdx.x & 31;
  int wv   = threadIdx.x >> 5;
  int ntile = blockIdx.x * (blockDim.x >> 5) + wv;
  int nbase = ntile * 16;
  int ml = lane & 15, g = lane >> 4;
  const float* a0p = A + (size_t)ml * K;
  const float* a1p = A + (size_t)(16+ml) * K;
  const float* wp  = W + (size_t)(nbase+ml) * K;
  v8f c0={}, c1={};
  for (int kc = 0; kc < K; kc += 32) {
    if (kc + 512 < K) __builtin_prefetch(wp + kc + 512, 0, 1);  // global_prefetch_b8
    int o0 = kc + g*8, o1 = kc + 16 + g*8;
    v16bf bf  = load_frag_contig(wp+o0,  wp+o1);
    v16bf af0 = load_frag_contig(a0p+o0, a0p+o1);
    v16bf af1 = load_frag_contig(a1p+o0, a1p+o1);
    c0 = wmma_bf16(af0, bf, c0);
    c1 = wmma_bf16(af1, bf, c1);
  }
  const float inv = rsqrtf(1.0f + 1e-5f);
  int n = nbase + (lane & 15);
  float bv = bias[n];
  float scale = 0.f, shift = 0.f;
  if (BN) { int c = n >> 7; scale = bng[c]*inv; shift = bnb[c]; }
  #pragma unroll
  for (int rr=0; rr<8; rr++){
    int m0 = (lane<16)?rr:(8+rr);
    int m1 = m0 + 16;
    float v0 = c0[rr]+bv, v1 = c1[rr]+bv;
    if (RELU){ v0=fmaxf(v0,0.f); v1=fmaxf(v1,0.f); }
    if (BN){
      v0 = (v0 + resid[(size_t)m0*N + n]) * scale + shift;
      v1 = (v1 + resid[(size_t)m1*N + n]) * scale + shift;
    }
    out[(size_t)m0*N + n] = v0;
    out[(size_t)m1*N + n] = v1;
  }
}

// ---------------------------------------------------------------------------
// Host orchestration.
// Input order: x | 8x(w1,b1,w2,b2,w3,b3) | 5x(wq,bq,wk,bk,wv,bv,wo,bo,
//   bn1g,bn1b,f1w,f1b,f2w,f2b,bn2g,bn2b) | f_w | f_b
// ---------------------------------------------------------------------------
extern "C" void kernel_launch(void* const* d_in, const int* in_sizes, int n_in,
                              void* d_out, int out_size, void* d_ws, size_t ws_size,
                              hipStream_t stream) {
  (void)in_sizes; (void)n_in; (void)out_size; (void)ws_size;
  const float* x = (const float*)d_in[0];
  static const int KS[8] = {256,128,64,32,16,8,4,2};
  ConvAll ca;
  for (int i=0;i<8;i++){
    ca.p[i].w1=(const float*)d_in[1+6*i+0];
    ca.p[i].b1=(const float*)d_in[1+6*i+1];
    ca.p[i].w2=(const float*)d_in[1+6*i+2];
    ca.p[i].b2=(const float*)d_in[1+6*i+3];
    ca.p[i].w3=(const float*)d_in[1+6*i+4];
    ca.p[i].b3=(const float*)d_in[1+6*i+5];
    ca.p[i].k = KS[i];
  }
  float* ws = (float*)d_ws;
  float* hA = ws;               // (B,C,L) current activations  (512 KB)
  float* hB = ws + 1*131072;    // post-BN1 activations
  float* qb = ws + 2*131072;    // q (B,H,L,D)
  float* kb = ws + 3*131072;    // k
  float* vb = ws + 4*131072;    // v
  float* ob = ws + 5*131072;    // attention output rows (4096 x 32)
  float* h1 = ws + 6*131072;    // FFN hidden (32 x 8192)  (1 MB)

  conv_kernel<<<BATCH, 256, 0, stream>>>(x, ca, hA);

  for (int i=0;i<5;i++){
    void* const* ap = d_in + 49 + 16*i;
    const float* wq  =(const float*)ap[0];  const float* bq  =(const float*)ap[1];
    const float* wk  =(const float*)ap[2];  const float* bk  =(const float*)ap[3];
    const float* wv  =(const float*)ap[4];  const float* bv  =(const float*)ap[5];
    const float* wo  =(const float*)ap[6];  const float* bo  =(const float*)ap[7];
    const float* b1g =(const float*)ap[8];  const float* b1b =(const float*)ap[9];
    const float* f1w =(const float*)ap[10]; const float* f1b =(const float*)ap[11];
    const float* f2w =(const float*)ap[12]; const float* f2b =(const float*)ap[13];
    const float* b2g =(const float*)ap[14]; const float* b2b =(const float*)ap[15];

    qkv_kernel<<<dim3(256,3), 32, 0, stream>>>(hA, wq,bq, wk,bk, wv,bv, qb,kb,vb);
    prob_attn_kernel<<<BATCH*NHEADS, SEQL, 0, stream>>>(qb, kb, vb, ob, i);
    attnproj_kernel<<<256, 32, 0, stream>>>(ob, wo, bo, hA, b1g, b1b, hB);
    // FFN1: (32,4096)x(8192,4096)^T -> relu -> h1
    wmma_gemm32<true,false><<<64, 256, 0, stream>>>(hB, f1w, f1b,
        nullptr, nullptr, nullptr, h1, 4096, 8192);
    // FFN2: (32,8192)x(4096,8192)^T -> relu -> +res2 -> BN2 -> hA
    wmma_gemm32<true,true><<<32, 256, 0, stream>>>(h1, f2w, f2b,
        hB, b2g, b2b, hA, 8192, 4096);
  }
  // Final projection: (32,4096)x(1024,4096)^T + f_b -> (32,1,1024)
  const float* fw = (const float*)d_in[129];
  const float* fb = (const float*)d_in[130];
  wmma_gemm32<false,false><<<8, 256, 0, stream>>>(hA, fw, fb,
      nullptr, nullptr, nullptr, (float*)d_out, 4096, 1024);
}
